// EdgeModel_49246095016466
// MI455X (gfx1250) — compile-verified, matrix operand order checked by reading, not run on hardware
//
#include <hip/hip_runtime.h>

typedef __attribute__((ext_vector_type(16))) __bf16 v16bf;
typedef __attribute__((ext_vector_type(8)))  float  v8f;
typedef __attribute__((ext_vector_type(4)))  float  f4;
typedef __attribute__((ext_vector_type(4)))  unsigned int u4;
typedef __attribute__((ext_vector_type(4)))  unsigned int ui4;
typedef __attribute__((ext_vector_type(8)))  int i8v;
typedef __attribute__((ext_vector_type(4)))  int i4v;

#define E_TOTAL   524288
#define NODE_IN   128
#define EDGE_IN   64
#define GLOBAL_IN 64
#define CAT_IN    384
#define HIDDEN    256
#define EDGE_OUT  64

#define W1T_ELEMS (HIDDEN * CAT_IN)      // 98304  bf16, blocked [12][256][32]
#define W2T_ELEMS (EDGE_OUT * HIDDEN)    // 16384  bf16, blocked [8][64][32]
#define W1T_BYTES (W1T_ELEMS * 2)        // 196608
#define W2T_BYTES (W2T_ELEMS * 2)        // 32768
#define WBLOB_BYTES (W1T_BYTES + W2T_BYTES)          // 229376 = 28672 x 8B
#define H_BYTES   (128 * HIDDEN * 2)     // 65536 : block-shared h [128][256] bf16
#define SMEM_BYTES (WBLOB_BYTES + H_BYTES)           // 294912 <= 320KB

#if defined(__gfx1250__) && __has_builtin(__builtin_amdgcn_tensor_load_to_lds) && \
    __has_builtin(__builtin_amdgcn_s_wait_tensorcnt)
#define USE_TDM 1
#else
#define USE_TDM 0
#endif

// ---- weight prep: fp32 -> bf16, K-blocked layout [kt][n][32] (WMMA B fragments
//      become contiguous 64B runs; LDS offsets collapse to small immediates) ----
__global__ void conv_w1(const float* __restrict__ W1, __bf16* __restrict__ W1t) {
    int idx = blockIdx.x * 256 + threadIdx.x;   // over 384*256, W1[k][n]
    int k = idx >> 8;
    int n = idx & 255;
    W1t[(size_t)(k >> 5) * (HIDDEN * 32) + n * 32 + (k & 31)] = (__bf16)W1[idx];
}

__global__ void conv_w2(const float* __restrict__ W2, __bf16* __restrict__ W2t) {
    int idx = blockIdx.x * 256 + threadIdx.x;   // over 256*64, W2[k][n]
    int k = idx >> 6;
    int n = idx & 63;
    W2t[(size_t)(k >> 5) * (EDGE_OUT * 32) + n * 32 + (k & 31)] = (__bf16)W2[idx];
}

struct RowPtrs { const float *s, *d, *a, *u; };

// branch-free source select for virtual concat x = [src | dest | edge_attr | u[batch]]
__device__ __forceinline__ const float* pick_ptr(int k, const RowPtrs& rp) {
    const float* p0 = rp.s + k;
    const float* p1 = rp.d + (k - 128);
    const float* p2 = rp.a + (k - 256);
    const float* p3 = rp.u + (k - 320);
    return k < 128 ? p0 : (k < 256 ? p1 : (k < 320 ? p2 : p3));
}

__device__ __forceinline__ v16bf lds_frag(const __bf16* p) {
    v16bf b;
    ((u4*)&b)[0] = *(const u4*)p;          // K [kb .. kb+7]
    ((u4*)&b)[1] = *(const u4*)(p + 16);   // K [kb+16 .. kb+23]
    return b;
}

__device__ __forceinline__ v16bf cvt_frag(f4 c0, f4 c1, f4 c2, f4 c3) {
    v16bf f;
    f[0] = (__bf16)c0.x; f[1] = (__bf16)c0.y; f[2]  = (__bf16)c0.z; f[3]  = (__bf16)c0.w;
    f[4] = (__bf16)c1.x; f[5] = (__bf16)c1.y; f[6]  = (__bf16)c1.z; f[7]  = (__bf16)c1.w;
    f[8] = (__bf16)c2.x; f[9] = (__bf16)c2.y; f[10] = (__bf16)c2.z; f[11] = (__bf16)c2.w;
    f[12] = (__bf16)c3.x; f[13] = (__bf16)c3.y; f[14] = (__bf16)c3.z; f[15] = (__bf16)c3.w;
    return f;
}

__global__ void __launch_bounds__(256)
edge_mlp_kernel(const float* __restrict__ src, const float* __restrict__ dst,
                const float* __restrict__ ea,  const float* __restrict__ u,
                const int*   __restrict__ batch,
                const __bf16* __restrict__ wblob_g,   // w1t then w2t, contiguous in ws
                const float* __restrict__ b1, const float* __restrict__ b2,
                float* __restrict__ out) {
    extern __shared__ char smem[];
    __bf16* w1t = (__bf16*)smem;                       // blocked [12][256][32]
    __bf16* w2t = (__bf16*)(smem + W1T_BYTES);         // blocked [8][64][32]
    __bf16* hsh = (__bf16*)(smem + WBLOB_BYTES);       // [128][256]

    const int tid  = threadIdx.x;
    const int wave = tid >> 5;

#if USE_TDM
    // ---- Tensor Data Mover: DMA the 224KB weight blob global->LDS (one wave) ----
    if (wave == 0) {
        const unsigned long long ga = (unsigned long long)(uintptr_t)wblob_g;
        // D# group 0: count=1, lds_addr=0, global_addr, type=2 ("image")
        ui4 g0 = { 1u,
                   0u,
                   (unsigned)(ga & 0xFFFFFFFFull),
                   (unsigned)((ga >> 32) & 0x01FFFFFFull) | (2u << 30) };
        // D# group 1: data_size=3 (8B), 2-D tensor 4096 x 7 (= 229376 B), tile = whole
        i8v g1 = { (int)(3u << 16),                       // wg_mask=0, data_size=8B
                   (int)((4096u & 0xFFFFu) << 16),        // tensor_dim0[15:0]
                   (int)((4096u >> 16) | (7u << 16)),     // dim0 hi | tensor_dim1 lo
                   (int)((4096u) << 16),                  // dim1 hi(0) | tile_dim0=4096
                   (int)(7u),                             // tile_dim1=7, tile_dim2=0
                   (int)(4096u),                          // tensor_dim0_stride[31:0]
                   0, 0 };
        i4v gz4 = { 0, 0, 0, 0 };
        i8v gz8 = { 0, 0, 0, 0, 0, 0, 0, 0 };
        // this toolchain's 6-arg form: (g0, g1, g2, g3, g4, cpol)
        __builtin_amdgcn_tensor_load_to_lds(g0, g1, gz4, gz4, gz8, 0);
        __builtin_amdgcn_s_wait_tensorcnt(0);
    }
#else
    {   // fallback: cooperative vector copy
        const u4* g = (const u4*)wblob_g;
        u4* l = (u4*)smem;
        const int n16 = WBLOB_BYTES / 16;
        for (int i = tid; i < n16; i += 256) l[i] = g[i];
    }
#endif
    __syncthreads();

    const int lane   = tid & 31;
    const int lane16 = lane & 15;
    const int half   = lane >> 4;     // K chunk select per WMMA 16-bit layout
    const int mblk   = wave >> 1;     // 4 M-blocks of 32 edges
    const int nhalf  = wave & 1;      // 2 N-halves of 128 hidden cols

    // ---------------- GEMM1: this wave: [32 x 384] x [384 x 128] ----------------
    const int r0 = blockIdx.x * 128 + mblk * 32 + lane16;  // A tile-0 row (edge)
    const int r1 = r0 + 16;                                // A tile-1 row
    const RowPtrs rp0{src + (size_t)r0 * NODE_IN, dst + (size_t)r0 * NODE_IN,
                      ea + (size_t)r0 * EDGE_IN, u + (size_t)batch[r0] * GLOBAL_IN};
    const RowPtrs rp1{src + (size_t)r1 * NODE_IN, dst + (size_t)r1 * NODE_IN,
                      ea + (size_t)r1 * EDGE_IN, u + (size_t)batch[r1] * GLOBAL_IN};

    v8f acc[2][8];
#pragma unroll
    for (int mi = 0; mi < 2; ++mi)
#pragma unroll
        for (int nt = 0; nt < 8; ++nt) acc[mi][nt] = (v8f){};

    f4 raw[8];  // single raw buffer: issue kt+1 loads, convert after inner loop
#define ISSUE_A(KT)                                                             \
    {                                                                           \
        const int kb_ = (KT) * 32 + half * 8;                                   \
        const float* q00 = pick_ptr(kb_,      rp0);                             \
        const float* q01 = pick_ptr(kb_ + 16, rp0);                             \
        const float* q10 = pick_ptr(kb_,      rp1);                             \
        const float* q11 = pick_ptr(kb_ + 16, rp1);                             \
        raw[0] = __builtin_nontemporal_load((const f4*)q00);                    \
        raw[1] = __builtin_nontemporal_load((const f4*)q00 + 1);                \
        raw[2] = __builtin_nontemporal_load((const f4*)q01);                    \
        raw[3] = __builtin_nontemporal_load((const f4*)q01 + 1);                \
        raw[4] = __builtin_nontemporal_load((const f4*)q10);                    \
        raw[5] = __builtin_nontemporal_load((const f4*)q10 + 1);                \
        raw[6] = __builtin_nontemporal_load((const f4*)q11);                    \
        raw[7] = __builtin_nontemporal_load((const f4*)q11 + 1);                \
    }

    ISSUE_A(0);
    v16bf a0 = cvt_frag(raw[0], raw[1], raw[2], raw[3]);
    v16bf a1 = cvt_frag(raw[4], raw[5], raw[6], raw[7]);

    // base pointer inside blocked W1: row (nhalf*128+lane16), chunk half*8
    const __bf16* wbase = w1t + (size_t)(nhalf * 128 + lane16) * 32 + half * 8;

#pragma unroll
    for (int kt = 0; kt < 12; ++kt) {
        if (kt < 11) ISSUE_A(kt + 1);                // HBM latency hidden by 16 WMMAs
        const __bf16* wkt = wbase + (size_t)kt * (HIDDEN * 32);  // one add per kt

        v16bf bcur = lds_frag(wkt);                  // per-nt offsets: nt*1024 B (imm)
#pragma unroll
        for (int nt = 0; nt < 8; ++nt) {
            v16bf bnext;
            if (nt < 7) bnext = lds_frag(wkt + (size_t)(nt + 1) * (16 * 32));
            acc[0][nt] = __builtin_amdgcn_wmma_f32_16x16x32_bf16(
                false, a0, false, bcur, (short)0, acc[0][nt], false, false);
            acc[1][nt] = __builtin_amdgcn_wmma_f32_16x16x32_bf16(
                false, a1, false, bcur, (short)0, acc[1][nt], false, false);
            bcur = bnext;
        }
        if (kt < 11) {
            a0 = cvt_frag(raw[0], raw[1], raw[2], raw[3]);
            a1 = cvt_frag(raw[4], raw[5], raw[6], raw[7]);
        }
    }
#undef ISSUE_A

    // bias + ReLU -> shared h [128][256] bf16
#pragma unroll
    for (int mi = 0; mi < 2; ++mi) {
#pragma unroll
        for (int nt = 0; nt < 8; ++nt) {
            const int col = nhalf * 128 + nt * 16 + lane16;
            const float bias = b1[col];
            __bf16* hp = hsh + (size_t)(mblk * 32 + mi * 16 + half * 8) * HIDDEN + col;
#pragma unroll
            for (int i = 0; i < 8; ++i) {
                float h = acc[mi][nt][i] + bias;
                h = h > 0.f ? h : 0.f;
                hp[(size_t)i * HIDDEN] = (__bf16)h;
            }
        }
    }
    __syncthreads();

    // ---------------- GEMM2: [16 x 256] x [256 x 64] per wave -------------------
    const int er = blockIdx.x * 128 + wave * 16;
    const __bf16* arow = hsh + (size_t)(wave * 16 + lane16) * HIDDEN + half * 8;
    const __bf16* w2base = w2t + (size_t)lane16 * 32 + half * 8;

    v8f acc2[4];
#pragma unroll
    for (int nt = 0; nt < 4; ++nt) acc2[nt] = (v8f){};

#pragma unroll
    for (int kt = 0; kt < 8; ++kt) {
        v16bf afrag = lds_frag(arow + kt * 32);              // imm offsets
        const __bf16* w2kt = w2base + (size_t)kt * (EDGE_OUT * 32);

        v16bf bcur = lds_frag(w2kt);
#pragma unroll
        for (int nt = 0; nt < 4; ++nt) {
            v16bf bnext;
            if (nt < 3) bnext = lds_frag(w2kt + (size_t)(nt + 1) * (16 * 32));
            acc2[nt] = __builtin_amdgcn_wmma_f32_16x16x32_bf16(
                false, afrag, false, bcur, (short)0, acc2[nt], false, false);
            bcur = bnext;
        }
    }

    // bias + NT store fp32 output
#pragma unroll
    for (int nt = 0; nt < 4; ++nt) {
        const float bias = b2[nt * 16 + lane16];
#pragma unroll
        for (int i = 0; i < 8; ++i) {
            __builtin_nontemporal_store(
                acc2[nt][i] + bias,
                &out[(size_t)(er + i + half * 8) * EDGE_OUT + nt * 16 + lane16]);
        }
    }
}

extern "C" void kernel_launch(void* const* d_in, const int* in_sizes, int n_in,
                              void* d_out, int out_size, void* d_ws, size_t ws_size,
                              hipStream_t stream) {
    const float* src   = (const float*)d_in[0];
    const float* dst   = (const float*)d_in[1];
    const float* ea    = (const float*)d_in[2];
    const float* u     = (const float*)d_in[3];
    const int*   batch = (const int*)d_in[4];
    const float* W1    = (const float*)d_in[5];
    const float* b1    = (const float*)d_in[6];
    const float* W2    = (const float*)d_in[7];
    const float* b2    = (const float*)d_in[8];
    float* out = (float*)d_out;

    __bf16* w1t = (__bf16*)d_ws;                           // blocked bf16 (196608 B)
    __bf16* w2t = (__bf16*)((char*)d_ws + W1T_BYTES);      // blocked bf16 (contiguous)

    conv_w1<<<W1T_ELEMS / 256, 256, 0, stream>>>(W1, w1t);
    conv_w2<<<W2T_ELEMS / 256, 256, 0, stream>>>(W2, w2t);

    (void)in_sizes; (void)n_in; (void)out_size; (void)ws_size;
    edge_mlp_kernel<<<E_TOTAL / 128, 256, SMEM_BYTES, stream>>>(
        src, dst, ea, u, batch, w1t, b1, b2, out);
}
